// Attention_13700945674736
// MI455X (gfx1250) — matrix operand order repaired and checked by reading, not compiled, not run on hardware
//
#include <hip/hip_runtime.h>

#define BATCH 32
#define SEQ   1024
#define DIM   384
#define HEADS 8
#define HD    48     // head dim
#define HDP   64     // padded head dim for QK^T
#define WIMG  64     // image width
#define LDW   456    // scores LDS row stride (floats), mult of 4, 16B-aligned rows
#define LDK   72     // K-band LDS row stride (bf16): 144B rows (16B mult)
#define LDV   456    // V-band LDS row stride (bf16): 912B rows (16B mult)
#define LDA_X 390    // x-tile LDS stride (bf16): odd-dword -> conflict-free
#define LDA_P 392    // AO-tile LDS stride (bf16): 16B mult for async B128

typedef __attribute__((ext_vector_type(16))) __bf16 v16bf;
typedef __attribute__((ext_vector_type(8)))  float  v8f;

union FragAB { v16bf v; unsigned int u[8]; __bf16 h[16]; };
union PairBF { __bf16 h[2]; unsigned int u; };

// ---- CDNA5 async global->LDS DMA (GLOBAL_LOAD_ASYNC_TO_LDS_B128) ----------
__device__ __forceinline__ void async_copy16(const void* g, void* l) {
  unsigned la = (unsigned)(unsigned long long)l;   // low 32 bits = LDS offset
  unsigned long long ga = (unsigned long long)g;
  asm volatile("global_load_async_to_lds_b128 %0, %1, off"
               :: "v"(la), "v"(ga) : "memory");
}
__device__ __forceinline__ void wait_async() {
  asm volatile("s_wait_asynccnt 0x0" ::: "memory");
}

// A-matrix 16x32 bf16 fragment from fp32 row-major source (global or LDS).
// lane: m = lane%16, half = lane/16 (K+8); VGPR g holds pair kk=(g<4?2g:16+2(g-4)).
__device__ __forceinline__ void loadA_f32(FragAB& fr, const float* base,
                                          int ld, int k0, int lane) {
  const int m = lane & 15, half = lane >> 4;
  const float* p = base + (size_t)m * ld + k0 + (half ? 8 : 0);
#pragma unroll
  for (int g = 0; g < 8; ++g) {
    const int kk = (g < 4) ? (2 * g) : (16 + 2 * (g - 4));
    float2 t = *(const float2*)(p + kk);
    fr.h[2 * g]     = (__bf16)t.x;   // native v_cvt, RNE
    fr.h[2 * g + 1] = (__bf16)t.y;
  }
}

// A-matrix 16x32 bf16 fragment from bf16 row-major source (global or LDS).
__device__ __forceinline__ void loadA_bf16(FragAB& fr, const __bf16* base,
                                           int ld, int k0, int lane) {
  const int m = lane & 15, half = lane >> 4;
  const __bf16* p = base + (size_t)m * ld + k0 + (half ? 8 : 0);
#pragma unroll
  for (int g = 0; g < 8; ++g) {
    const int kk = (g < 4) ? (2 * g) : (16 + 2 * (g - 4));
    fr.u[g] = *(const unsigned int*)(p + kk);
  }
}

// B-matrix 32x16 bf16 fragment; memory row n (output col) holds K contiguous.
// lane: n = lane%16, half adds K+8; VGPR g holds pair kk=(g<4?2g:16+2(g-4)).
__device__ __forceinline__ void loadB_bf16(FragAB& fr, const __bf16* base,
                                           int ld, int k0, int lane) {
  const int n = lane & 15, half = lane >> 4;
  const __bf16* p = base + (size_t)n * ld + k0 + (half ? 8 : 0);
#pragma unroll
  for (int g = 0; g < 8; ++g)
    fr.u[g] = *(const unsigned int*)(p + ((g < 4) ? (2 * g) : (16 + 2 * (g - 4))));
}

__device__ __forceinline__ v8f bwmma(const FragAB& a, const FragAB& b, v8f c) {
  return __builtin_amdgcn_wmma_f32_16x16x32_bf16(false, a.v, false, b.v,
                                                 (short)0, c, false, false);
}

// ---------------- kernel 1: weights -> transposed bf16 -------------------
__global__ void prep_w_k(const float* __restrict__ wqkv,
                         const float* __restrict__ wproj,
                         __bf16* __restrict__ Wt, __bf16* __restrict__ Wpt) {
  int idx = blockIdx.x * blockDim.x + threadIdx.x;
  const int NQ = 3 * DIM * DIM;  // 442368
  if (idx < NQ) {
    const int nn = idx / DIM, k = idx % DIM;
    Wt[idx] = (__bf16)wqkv[(size_t)k * (3 * DIM) + nn];
  } else {
    idx -= NQ;
    if (idx < DIM * DIM) {
      const int nn = idx / DIM, k = idx % DIM;
      Wpt[idx] = (__bf16)wproj[(size_t)k * DIM + nn];
    }
  }
}

// ---------------- kernel 2: zero d=48..63 pad of Q/K ---------------------
__global__ void pad_zero_k(__bf16* __restrict__ Qp, __bf16* __restrict__ Kp) {
  const int idx = blockIdx.x * blockDim.x + threadIdx.x;  // 0..1048575
  const int chunk = idx & 1;
  const int row = idx >> 1;
  __bf16* base = (row < BATCH * HEADS * SEQ) ? Qp : Kp;
  const int rl = row & (BATCH * HEADS * SEQ - 1);
  uint4* p = (uint4*)(base + (size_t)rl * HDP + HD + chunk * 8);
  *p = make_uint4(0u, 0u, 0u, 0u);
}

// QKV epilogue: scatter one 16x16 C tile into Q (scaled) / K / V^T bf16.
__device__ __forceinline__ void qkv_store(const v8f& acc, int nt, int mt, int lane,
                                          __bf16* __restrict__ Qp,
                                          __bf16* __restrict__ Kp,
                                          __bf16* __restrict__ Vt) {
  const int n = lane & 15, half = lane >> 4;
  const int which = nt / 24;            // uniform per wave
  const int ntp = nt - which * 24;      // uniform
  const int h = ntp / 3;                // uniform
  const int d = (ntp - h * 3) * 16 + n;
#pragma unroll
  for (int r = 0; r < 8; ++r) {
    const int sg = mt * 16 + r + 8 * half;
    const int b_ = sg >> 10, s = sg & 1023;
    const float v = acc[r];
    if (which == 0)
      Qp[((size_t)(b_ * HEADS + h) * SEQ + s) * HDP + d] =
          (__bf16)(v * 0.14433756729740643f);  // 1/sqrt(48)
    else if (which == 1)
      Kp[((size_t)(b_ * HEADS + h) * SEQ + s) * HDP + d] = (__bf16)v;
    else
      Vt[((size_t)(b_ * HEADS + h) * HD + d) * SEQ + s] = (__bf16)v;
  }
}

// ---------------- kernel 3: QKV GEMM (bf16 WMMA) -------------------------
// Block = 4 waves sharing one 16-row M tile (x converted to bf16 in LDS once).
// Each wave owns TWO adjacent 16-col N tiles: A fragment feeds 2 WMMAs.
__global__ __launch_bounds__(128) void qkv_gemm_k(
    const float* __restrict__ x, const __bf16* __restrict__ Wt,
    __bf16* __restrict__ Qp, __bf16* __restrict__ Kp, __bf16* __restrict__ Vt) {
  __shared__ __bf16 sA[16 * LDA_X];
  const int lane = threadIdx.x & 31, wave = threadIdx.x >> 5;
  const int mt = blockIdx.x;                        // 0..2047
  const int nt0 = (blockIdx.y * 4 + wave) * 2;      // 0,2,..,70
  const float* Ab = x + (size_t)mt * 16 * DIM;

  // stage + convert x tile (16x384 f32 -> bf16)
  for (int e2 = threadIdx.x; e2 < 16 * 192; e2 += 128) {  // 24 iters, uniform
    const int row = e2 / 192, c2 = e2 - row * 192;
    float2 t = *(const float2*)(Ab + (size_t)row * DIM + 2 * c2);
    PairBF cv;
    cv.h[0] = (__bf16)t.x;
    cv.h[1] = (__bf16)t.y;
    *(unsigned int*)(sA + row * LDA_X + 2 * c2) = cv.u;
  }
  __syncthreads();

  const __bf16* Bb0 = Wt + (size_t)nt0 * 16 * DIM;
  const __bf16* Bb1 = Bb0 + 16 * DIM;
  v8f acc0 = {0.f, 0.f, 0.f, 0.f, 0.f, 0.f, 0.f, 0.f};
  v8f acc1 = acc0;
  for (int k0 = 0; k0 < DIM; k0 += 32) {
    FragAB a, b0, b1;
    loadA_bf16(a, sA, LDA_X, k0, lane);
    loadB_bf16(b0, Bb0, DIM, k0, lane);
    loadB_bf16(b1, Bb1, DIM, k0, lane);
    acc0 = bwmma(a, b0, acc0);
    acc1 = bwmma(a, b1, acc1);
  }
  qkv_store(acc0, nt0, mt, lane, Qp, Kp, Vt);
  qkv_store(acc1, nt0 + 1, mt, lane, Qp, Kp, Vt);
}

// ---------------- kernel 4: fused banded attention -----------------------
// All 4 waves of a block share (b,h,image-row) => identical key band.
// K/V band DMA'd to LDS once per block via async-to-LDS B128.
__global__ __launch_bounds__(128) void attn_k(
    const __bf16* __restrict__ Qp, const __bf16* __restrict__ Kp,
    const __bf16* __restrict__ Vt, __bf16* __restrict__ AO) {
  extern __shared__ char smem_raw[];
  __bf16* sK = (__bf16*)smem_raw;                                  // 64512 B
  __bf16* sV = (__bf16*)(smem_raw + 448 * LDK * 2);                // 43776 B
  float*  sc = (float*)(smem_raw + 448 * LDK * 2 + HD * LDV * 2);  // 116736 B
  float*  linvA = (float*)(smem_raw + 448 * LDK * 2 + HD * LDV * 2 +
                           4 * 16 * LDW * 4);                      // 256 B

  const int lane = threadIdx.x & 31;
  const int wave = threadIdx.x >> 5;
  float* mysc = sc + wave * (16 * LDW);
  float* linv = linvA + wave * 16;

  const int tile = blockIdx.x * 4 + wave;  // 16-query wave tile
  const int bh = tile >> 6;
  const int qt = tile & 63;
  const int s0 = qt * 16;
  const int irow = s0 >> 6;                      // same for all 4 waves
  const int r0 = (irow - 3 > 0) ? irow - 3 : 0;
  const int r1 = (irow + 4 < 16) ? irow + 4 : 16;
  const int nk = (r1 - r0) * WIMG;               // 256..448, mult of 64
  const int tbase = r0 * WIMG;

  const __bf16* Kb = Kp + (size_t)bh * SEQ * HDP;
  const __bf16* Vb = Vt + (size_t)bh * HD * SEQ;

  // ---- stage K band: nk x 64 bf16, rows padded to LDK ----
  for (int c = threadIdx.x; c < nk * 8; c += 128) {  // uniform trip count
    const int row = c >> 3, part = c & 7;
    async_copy16(Kb + (size_t)(tbase + row) * HDP + part * 8,
                 sK + row * LDK + part * 8);
  }
  // ---- stage V band: 48 x nk bf16, rows padded to LDV ----
  const int nchunkV = nk >> 3;  // 16B chunks per V row
  for (int d = wave; d < HD; d += 4)
    for (int part = lane; part < nchunkV; part += 32)
      async_copy16(Vb + (size_t)d * SEQ + tbase + part * 8,
                   sV + d * LDV + part * 8);
  wait_async();
  __syncthreads();

  const int n = lane & 15, half = lane >> 4;
  const __bf16* Qb = Qp + (size_t)bh * SEQ * HDP + (size_t)s0 * HDP;
  FragAB qa0, qa1;
  loadA_bf16(qa0, Qb, HDP, 0, lane);
  loadA_bf16(qa1, Qb, HDP, 32, lane);

  const int jqb = s0 & 63;
  // scores -> LDS with analytic local-window bias (no mask memory traffic)
  for (int kt = 0; kt < (nk >> 4); ++kt) {
    FragAB kb0, kb1;
    loadB_bf16(kb0, sK + kt * 16 * LDK, LDK, 0, lane);
    loadB_bf16(kb1, sK + kt * 16 * LDK, LDK, 32, lane);
    v8f c = {0.f, 0.f, 0.f, 0.f, 0.f, 0.f, 0.f, 0.f};
    c = bwmma(qa0, kb0, c);
    c = bwmma(qa1, kb1, c);
    const int jt = ((tbase + kt * 16) & 63) + n;  // key col in image
#pragma unroll
    for (int r = 0; r < 8; ++r) {
      const int m = r + half * 8;
      const int dj = (jqb + m) - jt;
      float val = c[r];
      if (dj > 5 || dj < -5) val = -__builtin_inff();
      mysc[m * LDW + kt * 16 + n] = val;
    }
  }
  // softmax: two lanes per row, float4 LDS traffic
  {
    const int m = n;
    float* row = mysc + m * LDW;
    const int hn = nk >> 1;            // mult of 32
    float4* rv = (float4*)(row + half * hn);
    const int nq = hn >> 2;
    float mx = -__builtin_inff();
    for (int t = 0; t < nq; ++t) {
      float4 q = rv[t];
      mx = fmaxf(mx, fmaxf(fmaxf(q.x, q.y), fmaxf(q.z, q.w)));
    }
    mx = fmaxf(mx, __shfl_xor(mx, 16, 32));
    float sum = 0.f;
    for (int t = 0; t < nq; ++t) {
      float4 q = rv[t];
      q.x = __expf(q.x - mx); q.y = __expf(q.y - mx);
      q.z = __expf(q.z - mx); q.w = __expf(q.w - mx);
      rv[t] = q;
      sum += (q.x + q.y) + (q.z + q.w);
    }
    sum += __shfl_xor(sum, 16, 32);
    if (half == 0) linv[m] = 1.f / sum;
  }
  // P @ V  (A: LDS f32 -> bf16 frag, reused by 3 WMMAs; B: staged V band)
  v8f o0 = {0.f, 0.f, 0.f, 0.f, 0.f, 0.f, 0.f, 0.f};
  v8f o1 = o0, o2 = o0;
  for (int kc = 0; kc < nk; kc += 32) {
    FragAB p, v0, v1, v2;
    loadA_f32(p, mysc, LDW, kc, lane);
    loadB_bf16(v0, sV + 0 * LDV + kc, LDV, 0, lane);
    loadB_bf16(v1, sV + 16 * LDV + kc, LDV, 0, lane);
    loadB_bf16(v2, sV + 32 * LDV + kc, LDV, 0, lane);
    o0 = bwmma(p, v0, o0);
    o1 = bwmma(p, v1, o1);
    o2 = bwmma(p, v2, o2);
  }
  // epilogue: 1/rowsum scale, bf16 store into [B,S,C] (head-major cols)
  const int b = bh >> 3, h = bh & 7;
  const int cb = h * HD;
  float inv[8];
#pragma unroll
  for (int r = 0; r < 8; ++r) inv[r] = linv[r + half * 8];
#pragma unroll
  for (int r = 0; r < 8; ++r) {
    const size_t rowoff = ((size_t)b * SEQ + (s0 + r + 8 * half)) * DIM + cb + n;
    AO[rowoff + 0]  = (__bf16)(o0[r] * inv[r]);
    AO[rowoff + 16] = (__bf16)(o1[r] * inv[r]);
    AO[rowoff + 32] = (__bf16)(o2[r] * inv[r]);
  }
}

// ---------------- kernel 5: output projection + bias ---------------------
// AO M-tile staged once per block via async DMA; each wave owns 2 N tiles.
__global__ __launch_bounds__(128) void proj_gemm_k(
    const __bf16* __restrict__ AO, const __bf16* __restrict__ Wpt,
    const float* __restrict__ bias, float* __restrict__ out) {
  __shared__ __bf16 sA[16 * LDA_P];
  const int lane = threadIdx.x & 31, wave = threadIdx.x >> 5;
  const int mt = blockIdx.x;
  const int nt0 = (blockIdx.y * 4 + wave) * 2;  // 0,2,..,22
  const __bf16* Ab = AO + (size_t)mt * 16 * DIM;

  for (int c = threadIdx.x; c < 16 * 48; c += 128) {  // 6 iters, uniform
    const int row = c / 48, part = c - row * 48;
    async_copy16(Ab + (size_t)row * DIM + part * 8, sA + row * LDA_P + part * 8);
  }
  wait_async();
  __syncthreads();

  const __bf16* Bb0 = Wpt + (size_t)nt0 * 16 * DIM;
  const __bf16* Bb1 = Bb0 + 16 * DIM;
  v8f acc0 = {0.f, 0.f, 0.f, 0.f, 0.f, 0.f, 0.f, 0.f};
  v8f acc1 = acc0;
  for (int k0 = 0; k0 < DIM; k0 += 32) {
    FragAB a, b0, b1;
    loadA_bf16(a, sA, LDA_P, k0, lane);
    loadB_bf16(b0, Bb0, DIM, k0, lane);
    loadB_bf16(b1, Bb1, DIM, k0, lane);
    acc0 = bwmma(a, b0, acc0);
    acc1 = bwmma(a, b1, acc1);
  }
  const int n = lane & 15, half = lane >> 4;
#pragma unroll
  for (int tt = 0; tt < 2; ++tt) {
    const v8f acc = tt ? acc1 : acc0;
    const int col = (nt0 + tt) * 16 + n;
    const float bv = bias[col];
#pragma unroll
    for (int r = 0; r < 8; ++r) {
      const int sg = mt * 16 + r + 8 * half;
      out[(size_t)sg * DIM + col] = acc[r] + bv;
    }
  }
}

extern "C" void kernel_launch(void* const* d_in, const int* in_sizes, int n_in,
                              void* d_out, int out_size, void* d_ws, size_t ws_size,
                              hipStream_t stream) {
  (void)in_sizes; (void)n_in; (void)out_size; (void)ws_size;
  const float* x      = (const float*)d_in[0];
  const float* w_qkv  = (const float*)d_in[1];
  const float* w_proj = (const float*)d_in[2];
  const float* b_proj = (const float*)d_in[3];
  // d_in[4] (mask) unused: local window computed analytically.
  float* out = (float*)d_out;

  char* ws = (char*)d_ws;
  size_t off = 0;
  auto take = [&](size_t bytes) -> void* {
    void* p = ws + off;
    off += (bytes + 255) & ~(size_t)255;
    return p;
  };
  __bf16* Wt  = (__bf16*)take((size_t)3 * DIM * DIM * 2);
  __bf16* Wpt = (__bf16*)take((size_t)DIM * DIM * 2);
  __bf16* Qp  = (__bf16*)take((size_t)BATCH * HEADS * SEQ * HDP * 2);
  __bf16* Kp  = (__bf16*)take((size_t)BATCH * HEADS * SEQ * HDP * 2);
  __bf16* Vt  = (__bf16*)take((size_t)BATCH * HEADS * HD * SEQ * 2);
  __bf16* AO  = (__bf16*)take((size_t)BATCH * SEQ * DIM * 2);

  const int attn_lds = 448 * LDK * 2 + HD * LDV * 2 + 4 * 16 * LDW * 4 + 64 * 4;

  prep_w_k<<<2304, 256, 0, stream>>>(w_qkv, w_proj, Wt, Wpt);
  pad_zero_k<<<4096, 256, 0, stream>>>(Qp, Kp);
  qkv_gemm_k<<<dim3(2048, 9), 128, 0, stream>>>(x, Wt, Qp, Kp, Vt);
  attn_k<<<4096, 128, attn_lds, stream>>>(Qp, Kp, Vt, AO);
  proj_gemm_k<<<dim3(2048, 3), 128, 0, stream>>>(AO, Wpt, b_proj, out);
}
